// ConvMultiheadAttention2_71975061946809
// MI455X (gfx1250) — matrix operand order repaired and checked by reading, not compiled, not run on hardware
//
#include <hip/hip_runtime.h>
#include <hip/hip_bf16.h>

#define DNUM 1024
#define LNUM 1024
#define BNUM 4
#define HNUM 16
#define HDIM 64

typedef __bf16 bf16;
typedef __attribute__((ext_vector_type(16))) __bf16 v16bf;
typedef __attribute__((ext_vector_type(8)))  __bf16 v8bf;
typedef __attribute__((ext_vector_type(8)))  float  v8f;
typedef __attribute__((ext_vector_type(4)))  float  v4f;

union AFrag { v16bf v; v8bf h[2]; };

static __device__ __forceinline__ v8f wmma_bf16(const AFrag& a, const AFrag& b, v8f c) {
  // D = A(16x32) * B(32x16) + C, fp32 accumulate
  return __builtin_amdgcn_wmma_f32_16x16x32_bf16(false, a.v, false, b.v, (short)0, c, false, false);
}

// ---- DPP-based 16-lane butterfly reductions (stay within each half-wave) ----
template <int CTRL>
static __device__ __forceinline__ float dppf(float x) {
  int r = __builtin_amdgcn_mov_dpp(__builtin_bit_cast(int, x), CTRL, 0xF, 0xF, true);
  return __builtin_bit_cast(float, r);
}
static __device__ __forceinline__ float redmax16(float t) {
  t = fmaxf(t, dppf<0xB1>(t));    // xor 1 (quad_perm 1,0,3,2)
  t = fmaxf(t, dppf<0x4E>(t));    // xor 2 (quad_perm 2,3,0,1)
  t = fmaxf(t, dppf<0x141>(t));   // row_half_mirror (== xor 4 after above)
  t = fmaxf(t, dppf<0x140>(t));   // row_mirror      (== xor 8 after above)
  return t;
}
static __device__ __forceinline__ float redsum16(float t) {
  t += dppf<0xB1>(t);
  t += dppf<0x4E>(t);
  t += dppf<0x141>(t);
  t += dppf<0x140>(t);
  return t;
}

// ---------------- conversion kernels ----------------

__global__ __launch_bounds__(256) void cvt_act(const float* __restrict__ in,
                                               bf16* __restrict__ out, int n) {
  int i = blockIdx.x * 256 + threadIdx.x;
  if (i < n) out[i] = (bf16)in[i];
}

// w: fp32 [D_out][D_in][3]  ->  out: bf16 [3][D_out][D_in]
__global__ __launch_bounds__(256) void cvt_w(const float* __restrict__ in,
                                             bf16* __restrict__ out) {
  int i = blockIdx.x * 256 + threadIdx.x;   // over D*D
  if (i < DNUM * DNUM) {
    float t0 = in[i * 3 + 0], t1 = in[i * 3 + 1], t2 = in[i * 3 + 2];
    out[0 * DNUM * DNUM + i] = (bf16)t0;
    out[1 * DNUM * DNUM + i] = (bf16)t1;
    out[2 * DNUM * DNUM + i] = (bf16)t2;
  }
}

// ---------------- conv1d (K=3, same pad) as tiled WMMA GEMM ----------------
// Y[b, d_out, l] = sum_t sum_{d_in} W3[t][d_out][d_in] * X[b][l+t-1][d_in] + bias[d_out]
// X bf16 [B][L][D] (d contiguous).  OUT_MODE: 0 = bf16 [B][L][D],
// 1 = bf16 [B][D][L] (transposed), 2 = fp32 [B][L][D].
// Double-buffered software pipeline: global->regs for tile it+1 overlaps WMMA on tile it.

template <int OUT_MODE>
__global__ __launch_bounds__(256)
void conv_gemm(const bf16* __restrict__ X, const bf16* __restrict__ W3,
               const float* __restrict__ bias, void* __restrict__ Y) {
  __shared__ alignas(16) bf16 As[2][128 * 32];   // [m][k] (k contiguous)
  __shared__ alignas(16) bf16 Bs[2][128 * 32];   // [n][k] (k contiguous == global layout)

  const int b     = blockIdx.z;
  const int mtile = blockIdx.y * 128;         // d_out tile
  const int ntile = blockIdx.x * 128;         // l tile
  const int tid   = threadIdx.x;
  const int lane  = tid & 31;
  const int wid   = tid >> 5;
  const int wm    = (wid & 3) * 32;           // wave m offset (2 wmma tiles)
  const int wn    = (wid >> 2) * 64;          // wave n offset (4 wmma tiles)
  const int half  = lane >> 4;
  const int l16   = lane & 15;

  const int lrow = tid >> 1;                  // 0..127  (tile row loaded by this thread)
  const int lcol = (tid & 1) * 16;            // 0 or 16 (16 bf16 per thread)

  const v8f zf = {0.f, 0.f, 0.f, 0.f, 0.f, 0.f, 0.f, 0.f};
  v8f acc[2][4];
#pragma unroll
  for (int mm = 0; mm < 2; ++mm)
#pragma unroll
    for (int nn = 0; nn < 4; ++nn) acc[mm][nn] = zf;

  const bf16 zb = (bf16)0.0f;
  const v8bf zv = {zb, zb, zb, zb, zb, zb, zb, zb};

  const int NIT = 96;                         // 3 taps x 32 K-steps
  v8bf ra0, ra1, rb0, rb1;                    // staging registers

  auto gload = [&](int it) {
    const int t  = it >> 5;
    const int kd = (it & 31) * 32;
    const bf16* ga = W3 + (size_t)t * DNUM * DNUM +
                     (size_t)(mtile + lrow) * DNUM + kd + lcol;
    ra0 = *(const v8bf*)(ga);
    ra1 = *(const v8bf*)(ga + 8);
    const int lsrc = ntile + lrow + t - 1;    // shifted source column for tap t
    if (lsrc >= 0 && lsrc < LNUM) {
      const bf16* gb = X + ((size_t)b * LNUM + lsrc) * DNUM + kd + lcol;
      rb0 = *(const v8bf*)(gb);
      rb1 = *(const v8bf*)(gb + 8);
    } else {
      rb0 = zv;
      rb1 = zv;
    }
  };
  auto sstore = [&](int buf) {
    *(v8bf*)&As[buf][lrow * 32 + lcol]     = ra0;
    *(v8bf*)&As[buf][lrow * 32 + lcol + 8] = ra1;
    *(v8bf*)&Bs[buf][lrow * 32 + lcol]     = rb0;
    *(v8bf*)&Bs[buf][lrow * 32 + lcol + 8] = rb1;
  };

  gload(0);
  sstore(0);
  __syncthreads();

  for (int it = 0; it < NIT; ++it) {
    const int cur = it & 1;
    if (it + 1 < NIT) gload(it + 1);          // prefetch next tile into registers

    AFrag af[2], bf[4];
#pragma unroll
    for (int mm = 0; mm < 2; ++mm) {
      const bf16* pa = &As[cur][(wm + mm * 16 + l16) * 32];
      af[mm].h[0] = *(const v8bf*)(pa + half * 8);        // K = 8*half .. +7
      af[mm].h[1] = *(const v8bf*)(pa + 16 + half * 8);   // K = 16+8*half .. +7
    }
#pragma unroll
    for (int nn = 0; nn < 4; ++nn) {
      const bf16* pb = &Bs[cur][(wn + nn * 16 + l16) * 32];
      bf[nn].h[0] = *(const v8bf*)(pb + half * 8);        // K = 8*half .. +7
      bf[nn].h[1] = *(const v8bf*)(pb + 16 + half * 8);   // K = 16+8*half .. +7
    }
#pragma unroll
    for (int mm = 0; mm < 2; ++mm)
#pragma unroll
      for (int nn = 0; nn < 4; ++nn)
        acc[mm][nn] = wmma_bf16(af[mm], bf[nn], acc[mm][nn]);

    if (it + 1 < NIT) {
      sstore(1 - cur);                        // fill the other buffer
      __syncthreads();                        // includes s_wait_dscnt: safe reuse
    }
  }

  // bias + store (packed)
#pragma unroll
  for (int mm = 0; mm < 2; ++mm) {
    const int mbase = mtile + wm + mm * 16 + half * 8;    // C layout: M = r + 8*half
#pragma unroll
    for (int nn = 0; nn < 4; ++nn) {
      const int n = ntile + wn + nn * 16 + l16;           // C layout: N = lane&15
      if (OUT_MODE == 0) {
        v8bf pk;
#pragma unroll
        for (int r = 0; r < 8; ++r) pk[r] = (bf16)(acc[mm][nn][r] + bias[mbase + r]);
        *(v8bf*)&((bf16*)Y)[((size_t)b * LNUM + n) * DNUM + mbase] = pk;
      } else if (OUT_MODE == 1) {
#pragma unroll
        for (int r = 0; r < 8; ++r) {
          const float v = acc[mm][nn][r] + bias[mbase + r];
          ((bf16*)Y)[((size_t)b * DNUM + mbase + r) * LNUM + n] = (bf16)v;
        }
      } else {
        v4f lo, hi;
#pragma unroll
        for (int r = 0; r < 4; ++r) {
          lo[r] = acc[mm][nn][r] + bias[mbase + r];
          hi[r] = acc[mm][nn][r + 4] + bias[mbase + r + 4];
        }
        float* yp = &((float*)Y)[((size_t)b * LNUM + n) * DNUM + mbase];
        *(v4f*)(yp)     = lo;
        *(v4f*)(yp + 4) = hi;
      }
    }
  }
}

// ---------------- flash attention (bf16 WMMA, online softmax) ----------------
// Q,K bf16 [B][L][D]; VT bf16 [B][D][L]; AO bf16 [B][L][D]
__global__ __launch_bounds__(128)
void flash_attn(const bf16* __restrict__ Q, const bf16* __restrict__ K,
                const bf16* __restrict__ VT, const unsigned char* __restrict__ kpm,
                const float* __restrict__ amask, bf16* __restrict__ AO) {
  __shared__ alignas(16) bf16 Pl[4][16 * 32];

  const int b    = blockIdx.z;
  const int h    = blockIdx.y;
  const int wid  = threadIdx.x >> 5;
  const int lane = threadIdx.x & 31;
  const int half = lane >> 4;
  const int l16  = lane & 15;
  const int ib   = blockIdx.x * 64 + wid * 16;  // 16 query rows per wave
  const int hd   = h * HDIM;
  const float scale = 0.125f;                   // 64^-0.5
  const float NEGINF = -__builtin_inff();
  bf16* Pw = Pl[wid];

  // Q A-fragments (16 x 64 -> two 16x32 fragments), loaded once
  AFrag qf[2];
#pragma unroll
  for (int kt = 0; kt < 2; ++kt) {
    const bf16* pq = Q + ((size_t)b * LNUM + ib + l16) * DNUM + hd + kt * 32;
    qf[kt].h[0] = *(const v8bf*)(pq + half * 8);
    qf[kt].h[1] = *(const v8bf*)(pq + 16 + half * 8);
  }

  const v8f zf = {0.f, 0.f, 0.f, 0.f, 0.f, 0.f, 0.f, 0.f};
  v8f o[4];
#pragma unroll
  for (int dt = 0; dt < 4; ++dt) o[dt] = zf;
  float mrow[8], lsum[8];
#pragma unroll
  for (int r = 0; r < 8; ++r) { mrow[r] = NEGINF; lsum[r] = 0.f; }

  for (int jb = 0; jb < LNUM; jb += 32) {
    if (jb + 32 < LNUM)
      __builtin_prefetch(K + ((size_t)b * LNUM + jb + 32 + l16) * DNUM + hd, 0, 0);

    // S = scale * Q K^T for 32 keys (two 16x16 C tiles)
    v8f s[2];
#pragma unroll
    for (int sub = 0; sub < 2; ++sub) {
      s[sub] = zf;
      const bf16* pk = K + ((size_t)b * LNUM + jb + sub * 16 + l16) * DNUM + hd;
#pragma unroll
      for (int kt = 0; kt < 2; ++kt) {
        AFrag kf;
        const bf16* p = pk + kt * 32;
        kf.h[0] = *(const v8bf*)(p + half * 8);
        kf.h[1] = *(const v8bf*)(p + 16 + half * 8);
        s[sub] = wmma_bf16(qf[kt], kf, s[sub]);
      }
    }

    // masks + online softmax (reductions via DPP, no LDS traffic)
    const int j0 = jb + l16, j1 = j0 + 16;
    const bool kp0 = kpm[(size_t)b * LNUM + j0] != 0;
    const bool kp1 = kpm[(size_t)b * LNUM + j1] != 0;
#pragma unroll
    for (int r = 0; r < 8; ++r) {
      const int i = ib + half * 8 + r;                    // C layout row
      float s0 = s[0][r] * scale + amask[(size_t)i * LNUM + j0];
      float s1 = s[1][r] * scale + amask[(size_t)i * LNUM + j1];
      if (kp0) s0 = NEGINF;
      if (kp1) s1 = NEGINF;

      const float t  = redmax16(fmaxf(s0, s1));           // row max over 32 cols
      const float mn = fmaxf(mrow[r], t);
      const float corr = (mn > NEGINF) ? __expf(mrow[r] - mn) : 0.f;
      const float e0 = kp0 ? 0.f : __expf(s0 - mn);
      const float e1 = kp1 ? 0.f : __expf(s1 - mn);
      const float rs = redsum16(e0 + e1);                 // row sum
      lsum[r] = lsum[r] * corr + rs;
      mrow[r] = mn;
#pragma unroll
      for (int dt = 0; dt < 4; ++dt) o[dt][r] *= corr;

      // C-layout -> A-layout bounce through per-wave LDS (same-wave DS is in order)
      Pw[(r + half * 8) * 32 + l16]      = (bf16)e0;
      Pw[(r + half * 8) * 32 + 16 + l16] = (bf16)e1;
    }

    // P A-fragment (16x32)
    AFrag pf;
    const bf16* pp = &Pw[l16 * 32];
    pf.h[0] = *(const v8bf*)(pp + half * 8);
    pf.h[1] = *(const v8bf*)(pp + 16 + half * 8);

    // O += P * V  (4 d-tiles of 16)
#pragma unroll
    for (int dt = 0; dt < 4; ++dt) {
      AFrag vf;
      const bf16* pv = VT + ((size_t)b * DNUM + hd + dt * 16 + l16) * LNUM + jb;
      vf.h[0] = *(const v8bf*)(pv + half * 8);
      vf.h[1] = *(const v8bf*)(pv + 16 + half * 8);
      o[dt] = wmma_bf16(pf, vf, o[dt]);
    }
  }

  // normalize + store AO [B][L][D]
#pragma unroll
  for (int dt = 0; dt < 4; ++dt) {
#pragma unroll
    for (int r = 0; r < 8; ++r) {
      const float inv = 1.0f / lsum[r];
      const float v = o[dt][r] * inv;
      AO[((size_t)b * LNUM + ib + half * 8 + r) * DNUM + hd + dt * 16 + l16] = (bf16)v;
    }
  }
}

// ---------------- host launch ----------------

extern "C" void kernel_launch(void* const* d_in, const int* in_sizes, int n_in,
                              void* d_out, int out_size, void* d_ws, size_t ws_size,
                              hipStream_t stream) {
  const float* query = (const float*)d_in[0];
  const float* key   = (const float*)d_in[1];
  const float* value = (const float*)d_in[2];
  const unsigned char* kpm = (const unsigned char*)d_in[3];
  const float* amask = (const float*)d_in[4];
  const float* q_w = (const float*)d_in[5];
  const float* q_b = (const float*)d_in[6];
  const float* k_w = (const float*)d_in[7];
  const float* k_b = (const float*)d_in[8];
  const float* v_w = (const float*)d_in[9];
  const float* v_b = (const float*)d_in[10];
  const float* o_w = (const float*)d_in[11];
  const float* o_b = (const float*)d_in[12];

  char* ws = (char*)d_ws;
  size_t off = 0;
  auto take = [&](size_t bytes) -> char* {
    char* p = ws + off;
    off += (bytes + 255) & ~(size_t)255;
    return p;
  };
  const size_t WSZ = (size_t)3 * DNUM * DNUM * sizeof(bf16);
  const size_t ASZ = (size_t)BNUM * LNUM * DNUM * sizeof(bf16);

  bf16* W3Q = (bf16*)take(WSZ);
  bf16* W3K = (bf16*)take(WSZ);
  bf16* W3V = (bf16*)take(WSZ);
  bf16* W3O = (bf16*)take(WSZ);
  bf16* XQ  = (bf16*)take(ASZ);
  bf16* XK  = (bf16*)take(ASZ);
  bf16* XV  = (bf16*)take(ASZ);
  bf16* Qp  = (bf16*)take(ASZ);
  bf16* Kp  = (bf16*)take(ASZ);
  bf16* VTb = (bf16*)take(ASZ);   // [B][D][L]
  bf16* AO  = (bf16*)take(ASZ);
  (void)ws_size;

  // weight & activation conversion to bf16
  cvt_w<<<dim3(DNUM * DNUM / 256), 256, 0, stream>>>(q_w, W3Q);
  cvt_w<<<dim3(DNUM * DNUM / 256), 256, 0, stream>>>(k_w, W3K);
  cvt_w<<<dim3(DNUM * DNUM / 256), 256, 0, stream>>>(v_w, W3V);
  cvt_w<<<dim3(DNUM * DNUM / 256), 256, 0, stream>>>(o_w, W3O);
  const int nact = BNUM * LNUM * DNUM;
  cvt_act<<<dim3(nact / 256), 256, 0, stream>>>(query, XQ, nact);
  cvt_act<<<dim3(nact / 256), 256, 0, stream>>>(key,   XK, nact);
  cvt_act<<<dim3(nact / 256), 256, 0, stream>>>(value, XV, nact);

  // conv projections (WMMA GEMMs with 3-tap accumulation)
  dim3 cgrid(LNUM / 128, DNUM / 128, BNUM);
  conv_gemm<0><<<cgrid, 256, 0, stream>>>(XQ, W3Q, q_b, (void*)Qp);
  conv_gemm<0><<<cgrid, 256, 0, stream>>>(XK, W3K, k_b, (void*)Kp);
  conv_gemm<1><<<cgrid, 256, 0, stream>>>(XV, W3V, v_b, (void*)VTb);

  // attention
  flash_attn<<<dim3(LNUM / 64, HNUM, BNUM), 128, 0, stream>>>(Qp, Kp, VTb, kpm, amask, AO);

  // output conv -> fp32 [B][L][D]
  conv_gemm<2><<<cgrid, 256, 0, stream>>>(AO, W3O, o_b, d_out);
}
